// LiquidNeuralNetwork_24129126269644
// MI455X (gfx1250) — compile-verified, matrix operand order checked by reading, not run on hardware
//
#include <hip/hip_runtime.h>

// ---------------------------------------------------------------------------
// Liquid NN scan for gfx1250 (MI455X): bf16 WMMA GEMMs + f32 state update.
// B=32, T=512, D=H=O=1024, L=4.  Persistent kernel, 8 blocks x 8 waves = 64
// waves; wave w owns output columns [16w, 16w+16) and both 16-row tiles.
// - A panel (x_t ‖ h, 32x2048 bf16, 128 KB) staged in LDS once per block per
//   layer-step (CDNA5 WGP has 320 KB LDS), killing the 8x redundant L2 fetch.
// - f32 hidden state lives in VGPRs (layer loop unrolled, static indexing).
// - tanh uses the gfx1250 hardware transcendental (v_tanh_f32).
// ---------------------------------------------------------------------------

typedef __attribute__((ext_vector_type(16))) __bf16 v16bf;
typedef __attribute__((ext_vector_type(8)))  float  v8f;

#define B_   32
#define T_   512
#define D_   1024
#define H_   1024
#define L_   4
#define O_   1024
#define K2_  2048

#define NBLK  8
#define NTHR  256                  // 8 waves of 32
#define RS_   2056                 // LDS row stride (bf16): bank-conflict-free pad
#define LDSB_ (32 * RS_ * 2)       // 131,584 bytes of dynamic LDS

__device__ __forceinline__ v16bf ldfrag_g(const __bf16* p) {
    return *(const v16bf*)p;       // 32B/lane, 32B-aligned -> 2x global_load_b128
}

// LDS rows are 16B-aligned only (padded stride) -> two explicit b128 loads
__device__ __forceinline__ v16bf ldfrag_lds(const __bf16* p) {
    union { float4 f[2]; v16bf v; } u;
    u.f[0] = *(const float4*)p;
    u.f[1] = *(const float4*)(p + 8);
    return u.v;
}

__device__ __forceinline__ v8f wmma_bf16(v16bf a, v16bf b, v8f c) {
    return __builtin_amdgcn_wmma_f32_16x16x32_bf16(
        /*neg_a=*/false, a, /*neg_b=*/false, b,
        /*c_mod=*/(short)0, c, /*reuse_a=*/false, /*reuse_b=*/false);
}

// gfx1250 has a hardware transcendental tanh (TRANS32 class) — one op instead
// of the branchy ocml expansion that lands on the serialized recurrence path.
__device__ __forceinline__ float fast_tanh(float x) {
#if __has_builtin(__builtin_amdgcn_tanhf)
    return __builtin_amdgcn_tanhf(x);
#else
    float r;
    asm("v_tanh_f32 %0, %1" : "=v"(r) : "v"(x));
    return r;
#endif
}

__device__ __forceinline__ void grid_sync(unsigned* bar, unsigned* gen, unsigned nblk) {
    __threadfence();                 // release prior global writes (all waves/CUs)
    __syncthreads();
    if (threadIdx.x == 0) {
        unsigned g = __hip_atomic_load(gen, __ATOMIC_RELAXED, __HIP_MEMORY_SCOPE_AGENT);
        unsigned a = __hip_atomic_fetch_add(bar, 1u, __ATOMIC_ACQ_REL, __HIP_MEMORY_SCOPE_AGENT);
        if (a == nblk - 1u) {
            __hip_atomic_store(bar, 0u, __ATOMIC_RELAXED, __HIP_MEMORY_SCOPE_AGENT);
            __hip_atomic_fetch_add(gen, 1u, __ATOMIC_ACQ_REL, __HIP_MEMORY_SCOPE_AGENT);
        } else {
            while (__hip_atomic_load(gen, __ATOMIC_RELAXED, __HIP_MEMORY_SCOPE_AGENT) == g) {
                __builtin_amdgcn_s_sleep(1);
            }
        }
    }
    __syncthreads();
    __threadfence();                 // acquire: discard stale cached lines
}

// ---------------------------- prep kernels ---------------------------------

// x f32 [B,T,D] -> bf16 (same layout)
__global__ void lnn_prep_x(const float* __restrict__ x, __bf16* __restrict__ xb, int n) {
    int i = blockIdx.x * blockDim.x + threadIdx.x;
    if (i < n) xb[i] = (__bf16)x[i];
}

// WcT[l][n][k] = bf16( k<1024 ? W_in[l][k][n] : W_h[l][k-1024][n] )   (N-major, K contiguous)
__global__ void lnn_prep_wc(const float* __restrict__ W_in, const float* __restrict__ W_h,
                            __bf16* __restrict__ WcT) {
    long i = (long)blockIdx.x * blockDim.x + threadIdx.x;
    const long total = (long)L_ * H_ * K2_;
    if (i >= total) return;
    int k = (int)(i % K2_);
    long r = i / K2_;
    int n = (int)(r % H_);
    int l = (int)(r / H_);
    float v = (k < D_) ? W_in[((long)l * D_ + k) * H_ + n]
                       : W_h [((long)l * H_ + (k - D_)) * H_ + n];
    WcT[i] = (__bf16)v;
}

// WoT[n][k] = bf16(W_out[k][n])
__global__ void lnn_prep_wo(const float* __restrict__ W_out, __bf16* __restrict__ WoT) {
    long i = (long)blockIdx.x * blockDim.x + threadIdx.x;
    const long total = (long)O_ * H_;
    if (i >= total) return;
    int k = (int)(i % H_);
    int n = (int)(i / H_);
    WoT[i] = (__bf16)W_out[(long)k * O_ + n];
}

// combined bias, 1/tau, zero hb (bf16 double buffer), zero barrier
__global__ void lnn_prep_misc(const float* __restrict__ b_in, const float* __restrict__ b_h,
                              const float* __restrict__ tau,
                              float* __restrict__ bc, float* __restrict__ invtau,
                              __bf16* __restrict__ hb, unsigned* __restrict__ bar) {
    int i = blockIdx.x * blockDim.x + threadIdx.x;
    if (i < L_ * H_) {
        bc[i]     = b_in[i] + b_h[i];
        invtau[i] = 1.0f / tau[i];
    }
    if (i < 2 * L_ * B_ * H_)  hb[i] = (__bf16)0.0f;
    if (i < 128)               bar[i] = 0u;
}

// ---------------------------- persistent scan ------------------------------

__global__ __launch_bounds__(NTHR) void lnn_persistent(
    const __bf16* __restrict__ xb,      // [B,T,D] bf16
    const __bf16* __restrict__ WcT,     // [L][H][K2] bf16
    const __bf16* __restrict__ WoT,     // [O][H]  bf16
    const float*  __restrict__ bc,      // [L][H]
    const float*  __restrict__ invtau,  // [L][H]
    __bf16*       __restrict__ hb,      // [2][L][B][H] bf16 state (double buffer)
    const float*  __restrict__ b_out,   // [O]
    float*        __restrict__ out,     // [B][O]
    unsigned* bar, unsigned* gen)
{
    extern __shared__ __bf16 As[];     // [32][RS_] staged A panel (x_t ‖ h)

    const int lane  = threadIdx.x & 31;
    const int wid   = (int)(blockIdx.x * (NTHR / 32) + (threadIdx.x >> 5)); // 0..63
    const int lr    = lane & 15;           // row-within-half / column-within-tile
    const int khalf = (lane >> 4) << 4;    // 0 or 16: K-half held by this lane
    const int mbase = (lane >> 4) << 3;    // 0 or 8: D-tile row group
    const int n0    = wid << 4;            // first output column owned by wave
    const int ncol  = n0 + lr;

    // loop-invariant per-column scalars and register-resident f32 hidden state
    float bcv[L_], itv[L_];
    float hreg[L_][16];                // [l][r]: r<8 -> rows 0..15, r>=8 -> rows 16..31
    #pragma unroll
    for (int l = 0; l < L_; ++l) {
        bcv[l] = bc[l * H_ + ncol];
        itv[l] = invtau[l * H_ + ncol];
        #pragma unroll
        for (int r = 0; r < 16; ++r) hreg[l][r] = 0.0f;
    }

    for (int t = 0; t < T_; ++t) {
        const int cur = t & 1, old = cur ^ 1;
        #pragma unroll
        for (int l = 0; l < L_; ++l) {
            // A source, K in [0,1024): layer input (x_t for l==0, fresh h[l-1] else)
            const __bf16* srcA1;
            long strideA1;
            if (l == 0) { srcA1 = xb + (long)t * D_;                           strideA1 = (long)T_ * D_; }
            else        { srcA1 = hb + ((long)(cur * L_ + (l - 1)) * B_) * H_; strideA1 = H_; }
            // A source, K in [1024,2048): this layer's previous-step h
            const __bf16* srcA2 = hb + ((long)(old * L_ + l) * B_) * H_;

            // ---- stage A panel into LDS: 32 rows x 2048 bf16, 16B chunks ----
            #pragma unroll 4
            for (int c = threadIdx.x; c < 32 * 256; c += NTHR) {
                const int row = c >> 8;          // 0..31 (batch row)
                const int k   = (c & 255) << 3;  // 0..2047, step 8
                const __bf16* src = (k < D_) ? (srcA1 + row * strideA1 + k)
                                             : (srcA2 + (long)row * H_ + (k - D_));
                *(float4*)(&As[row * RS_ + k]) = *(const float4*)src;
            }
            __syncthreads();

            // ---- GEMM: acc[32x16] += A[32x2048] * B[2048x16] ----
            const __bf16* wrow = WcT + ((long)l * H_ + ncol) * K2_;
            v8f acc0 = {};   // batch rows 0..15
            v8f acc1 = {};   // batch rows 16..31

            #pragma unroll 2
            for (int kk = 0; kk < K2_; kk += 32) {
                const int ko = kk + khalf;
                v16bf a0 = ldfrag_lds(&As[lr * RS_ + ko]);
                v16bf a1 = ldfrag_lds(&As[(lr + 16) * RS_ + ko]);
                v16bf bw = ldfrag_g(wrow + ko);
                acc0 = wmma_bf16(a0, bw, acc0);
                acc1 = wmma_bf16(a1, bw, acc1);
            }

            // epilogue: hi = h + (tanh(acc + b_in + b_h) - h) / tau
            __bf16* hbw = hb + ((long)(cur * L_ + l) * B_) * H_;
            #pragma unroll
            for (int r = 0; r < 8; ++r) {
                float d0 = fast_tanh(acc0[r] + bcv[l]);
                float d1 = fast_tanh(acc1[r] + bcv[l]);
                float h0 = hreg[l][r];
                float h1 = hreg[l][r + 8];
                h0 += (d0 - h0) * itv[l];
                h1 += (d1 - h1) * itv[l];
                hreg[l][r]     = h0;
                hreg[l][r + 8] = h1;
                hbw[(long)(mbase + r)      * H_ + ncol] = (__bf16)h0;
                hbw[(long)(mbase + r + 16) * H_ + ncol] = (__bf16)h1;
            }

            grid_sync(bar, gen, NBLK);   // layers (and steps) are serial;
                                         // also fences LDS reuse across phases
        }
    }

    // final projection: out = h[L-1] @ W_out + b_out   (M=32, N=1024, K=1024)
    const __bf16* hfin  = hb + ((long)(((T_ - 1) & 1) * L_ + (L_ - 1)) * B_) * H_;
    const __bf16* worow = WoT + (long)ncol * H_;
    v8f acc0 = {};
    v8f acc1 = {};
    #pragma unroll 2
    for (int kk = 0; kk < H_; kk += 32) {
        const int ko = kk + khalf;
        v16bf a0 = ldfrag_g(hfin + (long)lr        * H_ + ko);
        v16bf a1 = ldfrag_g(hfin + (long)(lr + 16) * H_ + ko);
        v16bf bw = ldfrag_g(worow + ko);
        acc0 = wmma_bf16(a0, bw, acc0);
        acc1 = wmma_bf16(a1, bw, acc1);
    }
    const float bo = b_out[ncol];
    #pragma unroll
    for (int r = 0; r < 8; ++r) {
        const int m0 = mbase + r;
        out[(long)m0 * O_ + ncol]        = acc0[r] + bo;
        out[(long)(m0 + 16) * O_ + ncol] = acc1[r] + bo;
    }
}

// ---------------------------- launcher -------------------------------------

extern "C" void kernel_launch(void* const* d_in, const int* in_sizes, int n_in,
                              void* d_out, int out_size, void* d_ws, size_t ws_size,
                              hipStream_t stream) {
    (void)in_sizes; (void)n_in; (void)out_size; (void)ws_size;

    const float* x     = (const float*)d_in[0];
    const float* W_in  = (const float*)d_in[1];
    const float* b_in  = (const float*)d_in[2];
    const float* W_h   = (const float*)d_in[3];
    const float* b_h   = (const float*)d_in[4];
    const float* tau   = (const float*)d_in[5];
    const float* W_out = (const float*)d_in[6];
    const float* b_out = (const float*)d_in[7];
    float* out = (float*)d_out;

    char* ws = (char*)d_ws;
    size_t off = 0;
    auto carve = [&](size_t bytes) -> void* {
        void* p = ws + off;
        off = (off + bytes + 255) & ~(size_t)255;
        return p;
    };

    __bf16*   xb     = (__bf16*)  carve((size_t)B_ * T_ * D_ * 2);       // 32 MB
    __bf16*   WcT    = (__bf16*)  carve((size_t)L_ * H_ * K2_ * 2);      // 16 MB
    __bf16*   WoT    = (__bf16*)  carve((size_t)O_ * H_ * 2);            // 2 MB
    float*    bc     = (float*)   carve((size_t)L_ * H_ * 4);
    float*    invt   = (float*)   carve((size_t)L_ * H_ * 4);
    __bf16*   hb     = (__bf16*)  carve((size_t)2 * L_ * B_ * H_ * 2);
    unsigned* barrgn = (unsigned*)carve(512);
    unsigned* bar = barrgn;
    unsigned* gen = barrgn + 64;

    const int nx = B_ * T_ * D_;
    lnn_prep_x  <<<(nx + 255) / 256, 256, 0, stream>>>(x, xb, nx);
    const long nwc = (long)L_ * H_ * K2_;
    lnn_prep_wc <<<(unsigned)((nwc + 255) / 256), 256, 0, stream>>>(W_in, W_h, WcT);
    const long nwo = (long)O_ * H_;
    lnn_prep_wo <<<(unsigned)((nwo + 255) / 256), 256, 0, stream>>>(W_out, WoT);
    lnn_prep_misc<<<(2 * L_ * B_ * H_ + 255) / 256, 256, 0, stream>>>(
        b_in, b_h, tau, bc, invt, hb, bar);

    lnn_persistent<<<NBLK, NTHR, LDSB_, stream>>>(
        xb, WcT, WoT, bc, invt, hb, b_out, out, bar, gen);
}
